// CausalSelfAttention_2877628088484
// MI455X (gfx1250) — matrix operand order repaired and checked by reading, not compiled, not run on hardware
//
#include <hip/hip_runtime.h>

// Causal self-attention forward for MI455X (gfx1250, wave32, WMMA).
// bf16 tensor path (v_wmma_f32_16x16x32_bf16, f32 accum) with double-buffered
// async LDS staging (global_load_async_to_lds_b128 / ASYNCcnt): tile k+1 DMAs
// into the alternate LDS buffer while tile k's 16 WMMAs execute.
// ~103 GFLOP over ~50MB working set (fits in 192MB L2) -> matrix-engine bound.

#define TT 2048
#define CC 1024
#define NH 16
#define HD 64

typedef __attribute__((ext_vector_type(16))) __bf16 v16bf;
typedef __attribute__((ext_vector_type(8)))  float  v8f;

union V16 { v16bf v; uint4 u[2]; };

__device__ __forceinline__ unsigned short f2bf(float f) {
  union { float f; unsigned int u; } x; x.f = f;
  return (unsigned short)((x.u + 0x7FFFu + ((x.u >> 16) & 1u)) >> 16);  // RNE
}

__device__ __forceinline__ v8f v8f_zero() {
  v8f z = {0.f,0.f,0.f,0.f,0.f,0.f,0.f,0.f};
  return z;
}

// Async DMA: 16B/lane global -> LDS, tracked by ASYNCcnt (no VGPR data path).
__device__ __forceinline__ void async_ld_b128(unsigned lds_addr, unsigned byte_off,
                                              const void* base) {
  asm volatile("global_load_async_to_lds_b128 %0, %1, %2"
               :: "v"(lds_addr), "v"(byte_off), "s"(base) : "memory");
}
// In-order completion: waiting to <=N leaves only the N newest ops in flight.
__device__ __forceinline__ void async_wait0() {
  asm volatile("s_wait_asynccnt 0x0" ::: "memory");
}
__device__ __forceinline__ void async_wait8() {
  asm volatile("s_wait_asynccnt 0x8" ::: "memory");
}
__device__ __forceinline__ void async_wait4() {
  asm volatile("s_wait_asynccnt 0x4" ::: "memory");
}
__device__ __forceinline__ unsigned lds_off(const void* p) {
  return (unsigned)(uintptr_t)p;  // low 32 bits of generic ptr = LDS address
}

// ---------------------------------------------------------------- casts
__global__ __launch_bounds__(256) void cast_f32_bf16_v4(const float* __restrict__ s,
                                                        unsigned short* __restrict__ d,
                                                        int n4) {
  int i = blockIdx.x * blockDim.x + threadIdx.x;
  if (i < n4) {
    float4 f = ((const float4*)s)[i];
    union { unsigned short us[4]; uint2 u; } o;
    o.us[0] = f2bf(f.x); o.us[1] = f2bf(f.y);
    o.us[2] = f2bf(f.z); o.us[3] = f2bf(f.w);
    ((uint2*)d)[i] = o.u;
  }
}

// ---------------------------------------------------------------- GEMM
// C[M,N] = A[M,K] * B[N,K]^T, bf16 in, f32 accum.
// Block tile 128x128, 128 threads = 4 waves (2x2 wave grid), wave tile 64x64 =
// 4x4 WMMA frags: 16 wmma per 16 fragment ds_load_b128 per 32-K step.
// Double-buffered async staging; LDS row stride 40 elems (80B) keeps the
// 16-row b128 fragment reads on disjoint 4-bank groups (conflict-free).
// MODE 0: scatter bf16 q/k/v^T for attention (q pre-scaled by 1/sqrt(64)).
// MODE 1: f32 output, row stride CC.
template <int MODE>
__global__ __launch_bounds__(128) void gemm_bf16_wmma(
    const unsigned short* __restrict__ A,
    const unsigned short* __restrict__ Bw,
    int Kdim,
    unsigned short* __restrict__ qh,
    unsigned short* __restrict__ kh,
    unsigned short* __restrict__ vth,
    float* __restrict__ outf) {
  __shared__ __align__(16) unsigned short sA[2][128 * 40];
  __shared__ __align__(16) unsigned short sB[2][128 * 40];

  const int tid  = threadIdx.x;
  const int lane = tid & 31;
  const int wave = tid >> 5;     // 0..3
  const int wm   = wave & 1;     // row half of block tile
  const int wn   = wave >> 1;    // col half of block tile
  const int half = lane >> 4;
  const int idx  = lane & 15;
  const int m0   = blockIdx.y * 128;
  const int n0   = blockIdx.x * 128;

  v8f acc[4][4];
#pragma unroll
  for (int i = 0; i < 4; ++i)
#pragma unroll
    for (int j = 0; j < 4; ++j) acc[i][j] = v8f_zero();

  const unsigned ldsA0 = lds_off(&sA[0][0]);
  const unsigned ldsB0 = lds_off(&sB[0][0]);
  const unsigned BUFB  = 128 * 40 * 2;  // bytes per buffer
  const int r_st  = tid >> 2;    // 0..31
  const int cq_st = tid & 3;     // 16B chunk within 64B row

  // Issue one tile's A+B staging (8 async b128 per thread).
  auto stage = [&](int buf, int kk) {
    const unsigned ab = ldsA0 + (unsigned)buf * BUFB;
    const unsigned bb = ldsB0 + (unsigned)buf * BUFB;
#pragma unroll
    for (int j = 0; j < 4; ++j) {
      const int r = r_st + j * 32;
      async_ld_b128(ab + (unsigned)(r * 40 + cq_st * 8) * 2,
                    (unsigned)(((m0 + r) * Kdim + kk + cq_st * 8) * 2), A);
    }
#pragma unroll
    for (int j = 0; j < 4; ++j) {
      const int r = r_st + j * 32;
      async_ld_b128(bb + (unsigned)(r * 40 + cq_st * 8) * 2,
                    (unsigned)(((n0 + r) * Kdim + kk + cq_st * 8) * 2), Bw);
    }
  };

  const int nk = Kdim >> 5;
  stage(0, 0);  // preload tile 0
  for (int kb = 0; kb < nk; ++kb) {
    const int buf = kb & 1;
    if (kb + 1 < nk) {       // stream next tile behind this tile's compute
      stage(buf ^ 1, (kb + 1) << 5);
      async_wait8();         // only current tile (the 8 oldest) must be done
    } else {
      async_wait0();
    }
    __syncthreads();

    const unsigned short* cA = &sA[buf][0];
    const unsigned short* cB = &sB[buf][0];
    V16 a[4], b[4];
#pragma unroll
    for (int fm = 0; fm < 4; ++fm) {
      const int row = wm * 64 + fm * 16 + idx;
      // A 16-bit 16x32 layout: half0 -> K0..7/K16..23, half1 -> K8..15/K24..31
      a[fm].u[0] = *(const uint4*)&cA[row * 40 + half * 8];
      a[fm].u[1] = *(const uint4*)&cA[row * 40 + half * 8 + 16];
    }
#pragma unroll
    for (int fn = 0; fn < 4; ++fn) {
      const int row = wn * 64 + fn * 16 + idx;
      // B 16-bit 32x16 layout: half0 -> K0..7/K16..23, half1 -> K8..15/K24..31
      b[fn].u[0] = *(const uint4*)&cB[row * 40 + half * 8];
      b[fn].u[1] = *(const uint4*)&cB[row * 40 + half * 8 + 16];
    }
#pragma unroll
    for (int fn = 0; fn < 4; ++fn)
#pragma unroll
      for (int fm = 0; fm < 4; ++fm)
        acc[fm][fn] = __builtin_amdgcn_wmma_f32_16x16x32_bf16(
            false, a[fm].v, false, b[fn].v, (short)0, acc[fm][fn], false, false);
    __syncthreads();  // fragment reads retired -> next stage may overwrite buf
  }

  // Epilogue. C/D layout: vgpr r, lanes 0-15 -> M=r, lanes 16-31 -> M=8+r.
#pragma unroll
  for (int fm = 0; fm < 4; ++fm) {
#pragma unroll
    for (int fn = 0; fn < 4; ++fn) {
#pragma unroll
      for (int r = 0; r < 8; ++r) {
        const int m = m0 + wm * 64 + fm * 16 + (half << 3) + r;
        const int n = n0 + wn * 64 + fn * 16 + idx;
        const float val = acc[fm][fn][r];
        if (MODE == 0) {
          const int which = n >> 10;     // 0=q 1=k 2=v
          const int c = n & 1023;
          const int h = c >> 6;
          const int d = c & 63;
          const int bb = m >> 11;        // batch
          const int t = m & 2047;
          const int bh = bb * NH + h;
          if (which == 0)
            qh[((size_t)bh * TT + t) * HD + d] = f2bf(val * 0.125f);  // 1/sqrt(64)
          else if (which == 1)
            kh[((size_t)bh * TT + t) * HD + d] = f2bf(val);
          else
            vth[((size_t)bh * HD + d) * TT + t] = f2bf(val);          // V transposed
        } else {
          outf[(size_t)m * CC + n] = val;
        }
      }
    }
  }
}

// ---------------------------------------------------------------- attention
// Flash-attention core. 128 threads = 4 waves; wave w owns 16 query rows.
// 64-query block; uniform key-block trip count (causality by masking) keeps
// __syncthreads legal. 8 WMMAs per 32-key block; K/V^T double-buffered async.
__global__ __launch_bounds__(128) void attn_flash_wmma(
    const unsigned short* __restrict__ q,
    const unsigned short* __restrict__ k,
    const unsigned short* __restrict__ vt,
    unsigned short* __restrict__ y) {
  __shared__ __align__(16) unsigned short sK[2][32 * 72];   // keys x d, 144B stride
  __shared__ __align__(16) unsigned short sV[2][64 * 40];   // d x keys, 80B stride
  __shared__ __align__(16) unsigned short sP[4][16 * 40];   // per-wave P tile

  const int tid  = threadIdx.x;
  const int lane = tid & 31;
  const int wave = tid >> 5;
  const int half = lane >> 4;
  const int idx  = lane & 15;

  const int qbase = blockIdx.x * 64;
  const int bh    = blockIdx.y;
  const int qrow_base = qbase + wave * 16;

  const unsigned short* qp  = q  + (size_t)bh * TT * HD;
  const unsigned short* kp  = k  + (size_t)bh * TT * HD;
  const unsigned short* vtp = vt + (size_t)bh * HD * TT;

  const unsigned ldsK0 = lds_off(&sK[0][0]);
  const unsigned ldsV0 = lds_off(&sV[0][0]);
  const unsigned KBUFB = 32 * 72 * 2;
  const unsigned VBUFB = 64 * 40 * 2;

  // Issue one 32-key tile's K+V^T staging (4 async b128 per thread).
  auto stage_kv = [&](int buf, int k0) {
    const unsigned kb0 = ldsK0 + (unsigned)buf * KBUFB;
    const unsigned vb0 = ldsV0 + (unsigned)buf * VBUFB;
#pragma unroll
    for (int j = 0; j < 2; ++j) {
      const int r  = (tid >> 3) + j * 16;  // key row
      const int cq = tid & 7;
      async_ld_b128(kb0 + (unsigned)(r * 72 + cq * 8) * 2,
                    (unsigned)(((k0 + r) * HD + cq * 8) * 2), kp);
    }
#pragma unroll
    for (int j = 0; j < 2; ++j) {
      const int r  = (tid >> 2) + j * 32;  // d row
      const int cq = tid & 3;
      async_ld_b128(vb0 + (unsigned)(r * 40 + cq * 8) * 2,
                    (unsigned)((r * TT + k0 + cq * 8) * 2), vtp);
    }
  };

  // Q fragments (A layout, 16x64 = two 16x32 frags), loaded once from global.
  V16 qa[2];
  {
    const unsigned short* rowp = qp + (size_t)(qrow_base + idx) * HD;
#pragma unroll
    for (int kk = 0; kk < 2; ++kk) {
      qa[kk].u[0] = *(const uint4*)(rowp + kk * 32 + half * 8);
      qa[kk].u[1] = *(const uint4*)(rowp + kk * 32 + half * 8 + 16);
    }
  }

  float mrow[8], lrow[8];
  v8f o[4];
#pragma unroll
  for (int r = 0; r < 8; ++r) { mrow[r] = -1e30f; lrow[r] = 0.f; }
#pragma unroll
  for (int fn = 0; fn < 4; ++fn) o[fn] = v8f_zero();

  const int nblocks = (qbase >> 5) + 2;  // keys 0 .. qbase+63
  stage_kv(0, 0);                        // preload tile 0
  for (int kb = 0; kb < nblocks; ++kb) {
    const int k0  = kb * 32;
    const int buf = kb & 1;
    if (kb + 1 < nblocks) {
      stage_kv(buf ^ 1, k0 + 32);
      async_wait4();
    } else {
      async_wait0();
    }
    __syncthreads();

    const unsigned short* cK = &sK[buf][0];
    const unsigned short* cV = &sV[buf][0];

    // S = Q K^T for two 16-key sub-tiles.
    v8f s[2];
#pragma unroll
    for (int ns = 0; ns < 2; ++ns) {
      const int krow = ns * 16 + idx;
      V16 kf0, kf1;
      kf0.u[0] = *(const uint4*)&cK[krow * 72 + half * 8];
      kf0.u[1] = *(const uint4*)&cK[krow * 72 + half * 8 + 16];
      kf1.u[0] = *(const uint4*)&cK[krow * 72 + 32 + half * 8];
      kf1.u[1] = *(const uint4*)&cK[krow * 72 + 32 + half * 8 + 16];
      v8f sv = v8f_zero();
      sv = __builtin_amdgcn_wmma_f32_16x16x32_bf16(false, qa[0].v, false, kf0.v,
                                                   (short)0, sv, false, false);
      sv = __builtin_amdgcn_wmma_f32_16x16x32_bf16(false, qa[1].v, false, kf1.v,
                                                   (short)0, sv, false, false);
      s[ns] = sv;
    }

    // Online softmax; D-layout row = (half*8 + r), col = idx (per sub-tile).
#pragma unroll
    for (int r = 0; r < 8; ++r) {
      const int qrow = qrow_base + (half << 3) + r;
      float s0 = (k0 + idx      <= qrow) ? s[0][r] : -1e30f;
      float s1 = (k0 + 16 + idx <= qrow) ? s[1][r] : -1e30f;
      float mx = fmaxf(s0, s1);
#pragma unroll
      for (int off = 1; off < 16; off <<= 1)
        mx = fmaxf(mx, __shfl_xor(mx, off, 32));   // 16-lane group reduce
      const float mn   = fmaxf(mrow[r], mx);
      const float corr = __expf(mrow[r] - mn);
      const float p0 = __expf(s0 - mn);
      const float p1 = __expf(s1 - mn);
      float ps = p0 + p1;
#pragma unroll
      for (int off = 1; off < 16; off <<= 1)
        ps += __shfl_xor(ps, off, 32);
      lrow[r] = lrow[r] * corr + ps;
      mrow[r] = mn;
#pragma unroll
      for (int fn = 0; fn < 4; ++fn) o[fn][r] *= corr;
      const int pr = (half << 3) + r;
      sP[wave][pr * 40 + idx]      = f2bf(p0);
      sP[wave][pr * 40 + 16 + idx] = f2bf(p1);
    }
    // In-wave DS RAW: stores above feed cross-lane A-fragment loads below.
    asm volatile("s_wait_dscnt 0" ::: "memory");

    // P fragment (A layout) + O += P V.
    V16 pa;
    pa.u[0] = *(const uint4*)&sP[wave][idx * 40 + half * 8];
    pa.u[1] = *(const uint4*)&sP[wave][idx * 40 + half * 8 + 16];
#pragma unroll
    for (int fn = 0; fn < 4; ++fn) {
      const int drow = fn * 16 + idx;
      V16 vf;
      vf.u[0] = *(const uint4*)&cV[drow * 40 + half * 8];
      vf.u[1] = *(const uint4*)&cV[drow * 40 + half * 8 + 16];
      o[fn] = __builtin_amdgcn_wmma_f32_16x16x32_bf16(false, pa.v, false, vf.v,
                                                      (short)0, o[fn], false, false);
    }
    __syncthreads();  // all reads of this buffer retired before it is restaged
  }

  // Normalize and write y as bf16 in [b, t, h*64+d] layout for the out-proj GEMM.
  const int b = bh / NH;
  const int h = bh % NH;
#pragma unroll
  for (int fn = 0; fn < 4; ++fn) {
#pragma unroll
    for (int r = 0; r < 8; ++r) {
      const int t = qrow_base + (half << 3) + r;
      const int d = fn * 16 + idx;
      y[((size_t)b * TT + t) * CC + h * HD + d] = f2bf(o[fn][r] / lrow[r]);
    }
  }
}

// ---------------------------------------------------------------- launch
extern "C" void kernel_launch(void* const* d_in, const int* in_sizes, int n_in,
                              void* d_out, int out_size, void* d_ws, size_t ws_size,
                              hipStream_t stream) {
  const float* x     = (const float*)d_in[0];  // [2,2048,1024]
  const float* wqkv  = (const float*)d_in[1];  // [3072,1024]
  const float* wproj = (const float*)d_in[2];  // [1024,1024]
  float* out = (float*)d_out;                  // [2,2048,1024]

  char* ws = (char*)d_ws;
  const size_t MB = 1ull << 20;
  unsigned short* xb     = (unsigned short*)(ws + 0 * MB);   // 8MB
  unsigned short* wqkvb  = (unsigned short*)(ws + 8 * MB);   // 6MB
  unsigned short* wprojb = (unsigned short*)(ws + 14 * MB);  // 2MB
  unsigned short* qh     = (unsigned short*)(ws + 16 * MB);  // 8MB [b,h,t,d]
  unsigned short* kh     = (unsigned short*)(ws + 24 * MB);  // 8MB [b,h,t,d]
  unsigned short* vth    = (unsigned short*)(ws + 32 * MB);  // 8MB [b,h,d,t]
  unsigned short* yh     = (unsigned short*)(ws + 40 * MB);  // 8MB [b,t,c]

  const int nx = 2 * TT * CC;       // 4194304
  const int nq = 3 * CC * CC;       // 3145728
  const int np = CC * CC;           // 1048576
  cast_f32_bf16_v4<<<(nx / 4 + 255) / 256, 256, 0, stream>>>(x, xb, nx / 4);
  cast_f32_bf16_v4<<<(nq / 4 + 255) / 256, 256, 0, stream>>>(wqkv, wqkvb, nq / 4);
  cast_f32_bf16_v4<<<(np / 4 + 255) / 256, 256, 0, stream>>>(wproj, wprojb, np / 4);

  // qkv = x @ w_qkv^T : M=4096, N=3072, K=1024
  gemm_bf16_wmma<0><<<dim3(3072 / 128, 4096 / 128), 128, 0, stream>>>(
      xb, wqkvb, CC, qh, kh, vth, nullptr);

  // flash attention per (b,h) x 64-query block
  attn_flash_wmma<<<dim3(TT / 64, 2 * NH), 128, 0, stream>>>(qh, kh, vth, yh);

  // out = y @ w_proj^T : M=4096, N=1024, K=1024
  gemm_bf16_wmma<1><<<dim3(1024 / 128, 4096 / 128), 128, 0, stream>>>(
      yh, wprojb, CC, nullptr, nullptr, nullptr, out);
}
